// GCNNet_46677704573591
// MI455X (gfx1250) — matrix operand order repaired and checked by reading, not compiled, not run on hardware
//
#include <hip/hip_runtime.h>
#include <hip/hip_bf16.h>

typedef __attribute__((ext_vector_type(16))) _Float16 v16h;
typedef __attribute__((ext_vector_type(8)))  _Float16 v8h;
typedef __attribute__((ext_vector_type(8)))  float    v8f;

#define N_NODES 65536
#define N_EDGES 524288
#define N_GRAPHS 1024
#define NODES_PER_GRAPH 64
#define FD 78

static inline int P32(int x) { return (x + 31) & ~31; }

#define WMMA_F16(a, b, c) \
  __builtin_amdgcn_wmma_f32_16x16x32_f16(false, (a), false, (b), (short)0, (c), false, false)

// ---------------------------------------------------------------------------
// WMMA GEMM on pre-packed f16 operands.
//   A     : [M x Kp] row-major f16, Kp % 32 == 0, zero-padded K tail. M % 32 == 0.
//   Bt    : [Np x Kp] row-major f16 = B transposed, Np % 32 == 0, zero-padded.
//   C     : optional f32 out, row stride ldc, columns guarded by n < N.
//   Cself : optional second f32 out = C * dinvp[row]^2 (GCN self-loop term),
//           same ldc. Fused here to avoid re-reading C in a separate pass.
//   Ch    : optional packed f16 out, row stride ldch; cols >= N written as 0
//           so it can feed the next GEMM directly as a padded A.
// One wave computes a 32x32 C tile: 2 A fragments x 2 B fragments ->
// 4 x v_wmma_f32_16x16x32_f16 per 32-wide K step (8 b128 loads / 4 WMMA).
// K loop is software-pipelined (next-slice loads issued before the WMMAs
// consuming the current slice -> partial s_wait_loadcnt).
// Fragment layouts: f16 A/B both use contiguous k = 16*hi + s; f32 16x16 D per CDNA5 ISA 7.12.2.
// ---------------------------------------------------------------------------
__global__ __launch_bounds__(256) void wmma_gemm_f16(
    const _Float16* __restrict__ A, const _Float16* __restrict__ Bt,
    const float* __restrict__ bias, float* __restrict__ C,
    float* __restrict__ Cself, const float* __restrict__ dinvp,
    _Float16* __restrict__ Ch,
    int M, int N, int Kp, int ldc, int ldch, int relu, int tilesN)
{
  const int wave = threadIdx.x >> 5;
  const int lane = threadIdx.x & 31;
  const int tilesM = M >> 5;
  const int tile = blockIdx.x * (blockDim.x >> 5) + wave;
  if (tile >= tilesM * tilesN) return;      // wave-uniform: EXEC stays full
  const int tm = tile / tilesN;
  const int tn = tile - tm * tilesN;
  const int lo = lane & 15;
  const int hi = lane >> 4;

  // A: lane supplies row (tm*32 + rblk*16 + lo); K run starts at 16*hi.
  const _Float16* a0p = A + (size_t)(tm * 32 + lo) * Kp + 16 * hi;
  const _Float16* a1p = a0p + (size_t)16 * Kp;
  // B: lane supplies col (tn*32 + cblk*16 + lo); K run starts at 16*hi.
  const _Float16* b0p = Bt + (size_t)(tn * 32 + lo) * Kp + 16 * hi;
  const _Float16* b1p = b0p + (size_t)16 * Kp;

  auto ldA = [](const _Float16* p) -> v16h {
    v8h l = *(const v8h*)p;
    v8h h = *(const v8h*)(p + 8);
    return __builtin_shufflevector(l, h,
        0, 1, 2, 3, 4, 5, 6, 7, 8, 9, 10, 11, 12, 13, 14, 15);
  };

  v8f acc00 = {}, acc01 = {}, acc10 = {}, acc11 = {};

  // prologue: load first K slice
  v16h a0 = ldA(a0p);
  v16h a1 = ldA(a1p);
  v16h b0 = *(const v16h*)(b0p);
  v16h b1 = *(const v16h*)(b1p);

  for (int k0 = 0; k0 + 32 < Kp; k0 += 32) {
    // issue next-slice loads before consuming current slice
    v16h na0 = ldA(a0p + k0 + 32);
    v16h na1 = ldA(a1p + k0 + 32);
    v16h nb0 = *(const v16h*)(b0p + k0 + 32);
    v16h nb1 = *(const v16h*)(b1p + k0 + 32);
    acc00 = WMMA_F16(a0, b0, acc00);
    acc01 = WMMA_F16(a0, b1, acc01);
    acc10 = WMMA_F16(a1, b0, acc10);
    acc11 = WMMA_F16(a1, b1, acc11);
    a0 = na0; a1 = na1; b0 = nb0; b1 = nb1;
  }
  // epilogue: last slice
  acc00 = WMMA_F16(a0, b0, acc00);
  acc01 = WMMA_F16(a0, b1, acc01);
  acc10 = WMMA_F16(a1, b0, acc10);
  acc11 = WMMA_F16(a1, b1, acc11);

  const int n0 = tn * 32 + lo;              // D: lane holds col lo of each 16-block
  const int n1 = n0 + 16;
  const bool n0in = (n0 < N), n1in = (n1 < N);
  const float bv0 = (bias && n0in) ? bias[n0] : 0.0f;
  const float bv1 = (bias && n1in) ? bias[n1] : 0.0f;
#pragma unroll
  for (int r = 0; r < 8; ++r) {             // D vgpr r = row r + 8*hi
    const int mm0 = tm * 32 + 8 * hi + r;
    const int mm1 = mm0 + 16;
    float v00 = acc00[r] + bv0;  // rows [tm*32, +16), cols n0 block
    float v01 = acc01[r] + bv1;  // rows [tm*32, +16), cols n1 block
    float v10 = acc10[r] + bv0;  // rows [+16, +32),   cols n0 block
    float v11 = acc11[r] + bv1;  // rows [+16, +32),   cols n1 block
    if (relu) {
      v00 = fmaxf(v00, 0.0f); v01 = fmaxf(v01, 0.0f);
      v10 = fmaxf(v10, 0.0f); v11 = fmaxf(v11, 0.0f);
    }
    if (C) {
      if (n0in) { C[(size_t)mm0 * ldc + n0] = v00; C[(size_t)mm1 * ldc + n0] = v10; }
      if (n1in) { C[(size_t)mm0 * ldc + n1] = v01; C[(size_t)mm1 * ldc + n1] = v11; }
    }
    if (Cself) {                 // fused GCN self-loop term: h * dinv^2
      float d0 = dinvp[mm0]; d0 *= d0;
      float d1 = dinvp[mm1]; d1 *= d1;
      if (n0in) { Cself[(size_t)mm0 * ldc + n0] = v00 * d0;
                  Cself[(size_t)mm1 * ldc + n0] = v10 * d1; }
      if (n1in) { Cself[(size_t)mm0 * ldc + n1] = v01 * d0;
                  Cself[(size_t)mm1 * ldc + n1] = v11 * d1; }
    }
    if (Ch) {
      Ch[(size_t)mm0 * ldch + n0] = n0in ? (_Float16)v00 : (_Float16)0.0f;
      Ch[(size_t)mm1 * ldch + n0] = n0in ? (_Float16)v10 : (_Float16)0.0f;
      Ch[(size_t)mm0 * ldch + n1] = n1in ? (_Float16)v01 : (_Float16)0.0f;
      Ch[(size_t)mm1 * ldch + n1] = n1in ? (_Float16)v11 : (_Float16)0.0f;
    }
  }
}

// ---------------------------------------------------------------------------
// Packing kernels (f32 -> padded f16)
// ---------------------------------------------------------------------------
__global__ void pack_a_f16(const float* __restrict__ in, _Float16* __restrict__ out,
                           int K, int Kp, long totalP) {
  long i = (long)blockIdx.x * blockDim.x + threadIdx.x;
  if (i >= totalP) return;
  int r = (int)(i / Kp), k = (int)(i - (long)r * Kp);
  out[i] = (k < K) ? (_Float16)in[(long)r * K + k] : (_Float16)0.0f;
}

// Bt[n][k] = B[k][n], padded to [Np x Kp]
__global__ void pack_bt_f16(const float* __restrict__ B, _Float16* __restrict__ out,
                            int K, int N, int Kp, long totalP) {
  long i = (long)blockIdx.x * blockDim.x + threadIdx.x;
  if (i >= totalP) return;
  int n = (int)(i / Kp), k = (int)(i - (long)n * Kp);
  out[i] = (n < N && k < K) ? (_Float16)B[(long)k * N + n] : (_Float16)0.0f;
}

// ---------------------------------------------------------------------------
// Graph kernels
// ---------------------------------------------------------------------------
__global__ void zero_kernel(float* p, int n) {
  int i = blockIdx.x * blockDim.x + threadIdx.x;
  if (i < n) p[i] = 0.0f;
}

__global__ void deg_count_kernel(const int* __restrict__ dst, float* __restrict__ deg, int E) {
  int e = blockIdx.x * blockDim.x + threadIdx.x;
  if (e < E) atomicAdd(&deg[dst[e]], 1.0f);
}

__global__ void dinv_kernel(float* d, int n) {
  int i = blockIdx.x * blockDim.x + threadIdx.x;
  if (i < n) d[i] = rsqrtf(d[i] + 1.0f);
}

// coef[e] = dinv[src[e]] * dinv[dst[e]]  (same for all convs of a branch)
__global__ void coef_kernel(const int* __restrict__ src, const int* __restrict__ dst,
                            const float* __restrict__ dinv, float* __restrict__ coef,
                            int E) {
  int e = blockIdx.x * blockDim.x + threadIdx.x;
  if (e < E) coef[e] = dinv[src[e]] * dinv[dst[e]];
}

// out[dst,f] += coef[e] * h[src,f] over all edges
__global__ void edge_agg_kernel(const int* __restrict__ src,
                                const int* __restrict__ dst,
                                const float* __restrict__ coef,
                                const float* __restrict__ h,
                                float* __restrict__ out, int F, long total) {
  long i = (long)blockIdx.x * blockDim.x + threadIdx.x;
  if (i >= total) return;
  int e = (int)(i / F);
  int f = (int)(i - (long)e * F);
  int s = src[e], d = dst[e];
  atomicAdd(&out[(long)d * F + f], h[(long)s * F + f] * coef[e]);
}

// x = relu(x + bias) in place (f32) AND write zero-padded f16 copy [rows x Fp]
__global__ void bias_relu_pack_kernel(float* __restrict__ x,
                                      const float* __restrict__ bias,
                                      _Float16* __restrict__ xh,
                                      int F, int Fp, long totalP) {
  long i = (long)blockIdx.x * blockDim.x + threadIdx.x;
  if (i >= totalP) return;
  int node = (int)(i / Fp);
  int f = (int)(i - (long)node * Fp);
  if (f < F) {
    long j = (long)node * F + f;
    float v = fmaxf(x[j] + bias[f], 0.0f);
    x[j] = v;
    xh[i] = (_Float16)v;
  } else {
    xh[i] = (_Float16)0.0f;
  }
}

// pooled f16 [G x Fp] = max over 64 contiguous nodes (padded cols -> 0)
__global__ void pool_max_pack_kernel(const float* __restrict__ x,
                                     _Float16* __restrict__ out,
                                     int F, int Fp, long totalP) {
  long i = (long)blockIdx.x * blockDim.x + threadIdx.x;
  if (i >= totalP) return;
  int g = (int)(i / Fp);
  int f = (int)(i - (long)g * Fp);
  if (f >= F) { out[i] = (_Float16)0.0f; return; }
  const float* p = x + (long)g * NODES_PER_GRAPH * F + f;
  float m = -3.402823466e+38f;
#pragma unroll 4
  for (int j = 0; j < NODES_PER_GRAPH; ++j) m = fmaxf(m, p[(long)j * F]);
  out[i] = (_Float16)m;
}

// out[g] = h32[g,:] . w + b   (32 -> 1)
__global__ void head_out_kernel(const float* __restrict__ h32,
                                const float* __restrict__ w,
                                const float* __restrict__ b,
                                float* __restrict__ out) {
  int g = blockIdx.x * blockDim.x + threadIdx.x;
  if (g >= N_GRAPHS) return;
  float acc = b[0];
#pragma unroll
  for (int j = 0; j < 32; ++j) acc += h32[g * 32 + j] * w[j];
  out[g] = acc;
}

// ---------------------------------------------------------------------------
// Host orchestration
// ---------------------------------------------------------------------------
static inline unsigned blks(long total, int bs) {
  return (unsigned)((total + bs - 1) / bs);
}
static inline unsigned gemm_blocks(int M, int tilesN) {
  long tiles = (long)(M >> 5) * tilesN;
  return (unsigned)((tiles + 7) / 8);   // 8 waves per 256-thread block
}

extern "C" void kernel_launch(void* const* d_in, const int* in_sizes, int n_in,
                              void* d_out, int out_size, void* d_ws, size_t ws_size,
                              hipStream_t stream) {
  (void)in_sizes; (void)n_in; (void)out_size; (void)ws_size;

  // ---- workspace bump allocator (256B aligned) ----
  char* base = (char*)d_ws;
  size_t off = 0;
  auto alloc = [&](size_t bytes) {
    char* p = base + off;
    off += (bytes + 255) & ~(size_t)255;
    return (void*)p;
  };
  float*     dinv    = (float*)alloc((size_t)N_NODES * 4);
  float*     coef    = (float*)alloc((size_t)N_EDGES * 4);
  float*     hbuf    = (float*)alloc((size_t)N_NODES * (FD * 4) * 4);
  float*     abuf    = (float*)alloc((size_t)N_NODES * (FD * 4) * 4);
  _Float16*  aph     = (_Float16*)alloc((size_t)N_NODES * P32(FD * 4) * 2);
  _Float16*  xp      = (_Float16*)alloc((size_t)N_NODES * P32(FD) * 2);
  _Float16*  pooledh = (_Float16*)alloc((size_t)N_GRAPHS * P32(FD * 4) * 2);
  _Float16*  fc1h    = (_Float16*)alloc((size_t)N_GRAPHS * 1024 * 2);
  _Float16*  xch     = (_Float16*)alloc((size_t)N_GRAPHS * 256 * 2);
  _Float16*  h128h   = (_Float16*)alloc((size_t)N_GRAPHS * 128 * 2);
  float*     h32     = (float*)alloc((size_t)N_GRAPHS * 32 * 4);
  _Float16*  targeth = (_Float16*)alloc((size_t)N_GRAPHS * 1024 * 2);
  _Float16*  btbuf   = (_Float16*)alloc((size_t)1024 * 320 * 2); // max Np*Kp

  const float* target = (const float*)d_in[6];
  const int Fs[4] = {FD, FD, FD * 2, FD * 4};

  for (int b = 0; b < 2; ++b) {
    const float* x  = (const float*)d_in[b == 0 ? 0 : 3];
    const int*   ei = (const int*)  d_in[b == 0 ? 1 : 4];
    const int* src = ei;
    const int* dst = ei + N_EDGES;
    int p = 7 + b * 10;
    const float* convW[3] = {(const float*)d_in[p + 0], (const float*)d_in[p + 2],
                             (const float*)d_in[p + 4]};
    const float* convB[3] = {(const float*)d_in[p + 1], (const float*)d_in[p + 3],
                             (const float*)d_in[p + 5]};
    const float* FC1W = (const float*)d_in[p + 6];
    const float* FC1B = (const float*)d_in[p + 7];
    const float* FC2W = (const float*)d_in[p + 8];
    const float* FC2B = (const float*)d_in[p + 9];

    // degree -> dinv -> per-edge coefficients
    zero_kernel<<<blks(N_NODES, 256), 256, 0, stream>>>(dinv, N_NODES);
    deg_count_kernel<<<blks(N_EDGES, 256), 256, 0, stream>>>(dst, dinv, N_EDGES);
    dinv_kernel<<<blks(N_NODES, 256), 256, 0, stream>>>(dinv, N_NODES);
    coef_kernel<<<blks(N_EDGES, 256), 256, 0, stream>>>(src, dst, dinv, coef, N_EDGES);

    // pack X -> f16 [65536 x P32(78)=96]
    {
      long tot = (long)N_NODES * P32(FD);
      pack_a_f16<<<blks(tot, 256), 256, 0, stream>>>(x, xp, FD, P32(FD), tot);
    }

    const _Float16* curh = xp;
    for (int c = 0; c < 3; ++c) {
      int Kc = Fs[c], Fc = Fs[c + 1];
      int Kp = P32(Kc), Fp = P32(Fc);
      long nodeTotP = (long)N_NODES * Fp;
      long edgeTot = (long)N_EDGES * Fc;
      // pack W (Kc x Fc) -> Bt [Fp x Kp]
      long btTot = (long)Fp * Kp;
      pack_bt_f16<<<blks(btTot, 256), 256, 0, stream>>>(convW[c], btbuf, Kc, Fc, Kp, btTot);
      // H = cur @ W -> hbuf f32 (ldc = Fc); fused: abuf = H * dinv^2
      wmma_gemm_f16<<<gemm_blocks(N_NODES, Fp / 32), 256, 0, stream>>>(
          curh, btbuf, nullptr, hbuf, abuf, dinv, nullptr,
          N_NODES, Fc, Kp, Fc, 0, 0, Fp / 32);
      // edge scatter-add, then bias+relu (+ packed f16 copy)
      edge_agg_kernel<<<blks(edgeTot, 256), 256, 0, stream>>>(
          src, dst, coef, hbuf, abuf, Fc, edgeTot);
      bias_relu_pack_kernel<<<blks(nodeTotP, 256), 256, 0, stream>>>(
          abuf, convB[c], aph, Fc, Fp, nodeTotP);
      curh = aph;
    }

    // global max pool -> pooled f16 [1024 x 320]
    {
      int F = FD * 4, Fp = P32(FD * 4);
      long tot = (long)N_GRAPHS * Fp;
      pool_max_pack_kernel<<<blks(tot, 256), 256, 0, stream>>>(abuf, pooledh, F, Fp, tot);
    }

    // fc1: [1024,312]@[312,1024]+b, relu -> fc1h f16 [1024x1024]
    {
      int K = FD * 4, N = 1024, Kp = P32(K), Np = P32(N);
      long btTot = (long)Np * Kp;
      pack_bt_f16<<<blks(btTot, 256), 256, 0, stream>>>(FC1W, btbuf, K, N, Kp, btTot);
      wmma_gemm_f16<<<gemm_blocks(N_GRAPHS, Np / 32), 256, 0, stream>>>(
          pooledh, btbuf, FC1B, nullptr, nullptr, nullptr, fc1h,
          N_GRAPHS, N, Kp, 0, Np, 1, Np / 32);
    }
    // fc2: [1024,1024]@[1024,64]+b -> concat cols [b*64, b*64+64) of xch
    {
      int K = 1024, N = 64, Kp = 1024, Np = 64;
      long btTot = (long)Np * Kp;
      pack_bt_f16<<<blks(btTot, 256), 256, 0, stream>>>(FC2W, btbuf, K, N, Kp, btTot);
      wmma_gemm_f16<<<gemm_blocks(N_GRAPHS, Np / 32), 256, 0, stream>>>(
          fc1h, btbuf, FC2B, nullptr, nullptr, nullptr, xch + b * 64,
          N_GRAPHS, N, Kp, 0, 256, 0, Np / 32);
    }
  }

  // protein stream: xt = target @ xt_w + xt_b -> concat cols 128..255
  {
    const float* XTW = (const float*)d_in[27];
    const float* XTB = (const float*)d_in[28];
    int K = 1000, N = 128, Kp = P32(K), Np = 128;
    long at = (long)N_GRAPHS * Kp;
    pack_a_f16<<<blks(at, 256), 256, 0, stream>>>(target, targeth, K, Kp, at);
    long btTot = (long)Np * Kp;
    pack_bt_f16<<<blks(btTot, 256), 256, 0, stream>>>(XTW, btbuf, K, N, Kp, btTot);
    wmma_gemm_f16<<<gemm_blocks(N_GRAPHS, Np / 32), 256, 0, stream>>>(
        targeth, btbuf, XTB, nullptr, nullptr, nullptr, xch + 128,
        N_GRAPHS, N, Kp, 0, 256, 0, Np / 32);
  }

  // head
  {
    const float* F1W = (const float*)d_in[29];
    const float* F1B = (const float*)d_in[30];
    const float* F2W = (const float*)d_in[31];
    const float* F2B = (const float*)d_in[32];
    const float* OW  = (const float*)d_in[33];
    const float* OB  = (const float*)d_in[34];
    // relu(xc @ fc1 + b): [1024,256]@[256,128] -> h128h f16
    {
      long btTot = (long)128 * 256;
      pack_bt_f16<<<blks(btTot, 256), 256, 0, stream>>>(F1W, btbuf, 256, 128, 256, btTot);
      wmma_gemm_f16<<<gemm_blocks(N_GRAPHS, 4), 256, 0, stream>>>(
          xch, btbuf, F1B, nullptr, nullptr, nullptr, h128h,
          N_GRAPHS, 128, 256, 0, 128, 1, 4);
    }
    // relu(h128 @ fc2 + b): [1024,128]@[128,32] -> h32 f32
    {
      long btTot = (long)32 * 128;
      pack_bt_f16<<<blks(btTot, 256), 256, 0, stream>>>(F2W, btbuf, 128, 32, 128, btTot);
      wmma_gemm_f16<<<gemm_blocks(N_GRAPHS, 1), 256, 0, stream>>>(
          h128h, btbuf, F2B, h32, nullptr, nullptr, nullptr,
          N_GRAPHS, 32, 128, 32, 0, 1, 1);
    }
    head_out_kernel<<<blks(N_GRAPHS, 256), 256, 0, stream>>>(h32, OW, OB, (float*)d_out);
  }
}
